// SharedDeformConv_11484742549658
// MI455X (gfx1250) — matrix operand order repaired and checked by reading, not compiled, not run on hardware
//
#include <hip/hip_runtime.h>
#include <hip/hip_bf16.h>

// ---------------- types ----------------
typedef __bf16 bf16_t;
typedef bf16_t v16bf __attribute__((ext_vector_type(16)));
typedef float  v8f   __attribute__((ext_vector_type(8)));

struct FragU { uint4 lo; uint4 hi; };           // 32 bytes, matches v16bf

union V8FU { v8f v; float4 f4[2]; float f[8]; };

// round-to-nearest-even float -> bf16 bits
static __device__ __forceinline__ unsigned short f2bf(float f) {
    unsigned u = __float_as_uint(f);
    unsigned r = ((u >> 16) & 1u) + 0x7FFFu;
    u += r;
    return (unsigned short)(u >> 16);
}

// ---------------- problem constants ----------------
#define N_B   2
#define CIN   256
#define H_IN  132
#define W_IN  132
#define PLANE (H_IN * W_IN)       // 17424
#define HO    128
#define WO    128
#define HOWO  (HO * WO)           // 16384
#define GRP   4
#define CG    64
#define KTAP  9
#define COUT  256
#define OCH   72                  // offset channels

// stage-1 GEMM: Kdim per 32-ch block = 32*9 = 288 -> 9 kslices of 32, 5 ntiles (80>=72)
// stage-2 GEMM: Kdim per group = 64*9 = 576 -> 18 kslices of 32, 16 ntiles (256)

// ws layout (dwords)
#define PACK1_DW 0
#define PACK1_SZ (8 * 9 * 5 * 256)        //  92160 dwords
#define PACK2_DW (PACK1_DW + PACK1_SZ)
#define PACK2_SZ (4 * 18 * 16 * 256)      // 294912 dwords
#define OFFS_DW  (PACK2_DW + PACK2_SZ)    // 387072
#define OFFS_SZ  (N_B * OCH * HOWO)       // 2359296 dwords

// ======================================================================
// Pack offset_w (72,256,3,3) into WMMA-B fragments.
// tile = ((cblock*9 + kslice)*5 + ntile); dword index inside tile = lane*8 + v
// value pair: K0 = ks*32 + (lane>>4)*16 + 2v ; N = ntile*16 + (lane&15)
// reduction index K within a 32-channel block: K = c_local*9 + kk
// ======================================================================
__global__ __launch_bounds__(256) void pack_offw_kernel(
    const float* __restrict__ w, unsigned* __restrict__ dst) {
    int t = blockIdx.x * 256 + threadIdx.x;          // < 92160
    int v    = t & 7;
    int lane = (t >> 3) & 31;
    int tile = t >> 8;
    int nt = tile % 5;
    int ks = (tile / 5) % 9;
    int cb = tile / 45;
    int N  = nt * 16 + (lane & 15);
    int K0 = ks * 32 + (lane >> 4) * 16 + 2 * v;
    unsigned bits[2];
#pragma unroll
    for (int i = 0; i < 2; ++i) {
        int K  = K0 + i;
        int cl = K / 9;
        int kk = K - cl * 9;
        float val = (N < OCH) ? w[((N * CIN) + cb * 32 + cl) * 9 + kk] : 0.0f;
        bits[i] = f2bf(val);
    }
    dst[t] = bits[0] | (bits[1] << 16);
}

// ======================================================================
// Pack deform_w (256,256,3,3) into WMMA-B fragments per group.
// tile = ((g*18 + kslice)*16 + ntile); K = c*9 + kk within group (576)
// ======================================================================
__global__ __launch_bounds__(256) void pack_dw_kernel(
    const float* __restrict__ w, unsigned* __restrict__ dst) {
    int t = blockIdx.x * 256 + threadIdx.x;          // < 294912
    int v    = t & 7;
    int lane = (t >> 3) & 31;
    int tile = t >> 8;
    int nt = tile & 15;
    int ks = (tile >> 4) % 18;
    int g  = (tile >> 4) / 18;
    int N  = nt * 16 + (lane & 15);
    int K0 = ks * 32 + (lane >> 4) * 16 + 2 * v;
    unsigned bits[2];
#pragma unroll
    for (int i = 0; i < 2; ++i) {
        int K  = K0 + i;
        int c  = K / 9;
        int kk = K - c * 9;
        float val = w[((N * CIN) + g * CG + c) * 9 + kk];
        bits[i] = f2bf(val);
    }
    dst[t] = bits[0] | (bits[1] << 16);
}

// ======================================================================
// Stage 1: dilated 3x3 conv -> offsets, as WMMA GEMM.
// grid = N*HO*(WO/16) = 2048 blocks, 160 threads (5 waves, 1 ntile each).
// LDS A: 16 rows x 288 bf16, row stride 148 dwords (mult of 4, banks distinct)
// ======================================================================
#define S1_STRIDE_DW 148
__global__ __launch_bounds__(160) void offset_conv_kernel(
    const float* __restrict__ x, const unsigned* __restrict__ pack1,
    const float* __restrict__ offb, float* __restrict__ offs_out) {
    int bid = blockIdx.x;
    int wo0 = (bid & 7) * 16;
    int ho  = (bid >> 3) & 127;
    int n   = bid >> 10;
    int tid  = threadIdx.x;
    int lane = tid & 31;
    int wv   = tid >> 5;                  // 0..4 -> ntile

    __shared__ unsigned sA[16 * S1_STRIDE_DW];

    v8f acc = {};

    for (int cb = 0; cb < 8; ++cb) {
        __syncthreads();
        // im2col fill: A[p][c_local*9+kk] = x[n, cb*32+c_local, ho+2ky, wo0+p+2kx]
        for (int e = tid; e < 16 * 288; e += 160) {
            int p   = e & 15;
            int col = e >> 4;                 // 0..287
            int cl  = col / 9;
            int kk  = col - cl * 9;
            int ky  = kk / 3;
            int kx  = kk - ky * 3;
            float v = x[((size_t)(n * CIN + cb * 32 + cl) * H_IN + (ho + 2 * ky)) * W_IN
                        + (wo0 + p + 2 * kx)];
            ((unsigned short*)sA)[p * (S1_STRIDE_DW * 2) + col] = f2bf(v);
        }
        __syncthreads();

        int row = lane & 15;
        int kbd = (lane >> 4) * 8;            // dword offset of lane's K-half
        for (int ks = 0; ks < 9; ++ks) {
            const unsigned* ap = sA + row * S1_STRIDE_DW + ks * 16 + kbd;
            FragU fa; fa.lo = *(const uint4*)ap; fa.hi = *(const uint4*)(ap + 4);
            v16bf a = __builtin_bit_cast(v16bf, fa);
            const unsigned* bp = pack1 + ((cb * 9 + ks) * 5 + wv) * 256 + lane * 8;
            FragU fb; fb.lo = *(const uint4*)bp; fb.hi = *(const uint4*)(bp + 4);
            v16bf b = __builtin_bit_cast(v16bf, fb);
            acc = __builtin_amdgcn_wmma_f32_16x16x32_bf16(
                false, a, false, b, (short)0, acc, false, false);
        }
    }

    // store offsets (+bias); C layout: VGPR r, lanes<16 -> M=r, lanes>=16 -> M=8+r
    int ncol = lane & 15;
    int cout = wv * 16 + ncol;
    if (cout < OCH) {
        float bias = offb[cout];
        V8FU u; u.v = acc;
        size_t base = (((size_t)n * OCH + cout) * HO + ho) * WO + wo0 + (lane >> 4) * 8;
#pragma unroll
        for (int r = 0; r < 8; ++r) offs_out[base + r] = u.f[r] + bias;
    }
}

// ======================================================================
// Stage 2: deformable conv main GEMM.
// grid = 2048 blocks, 256 threads (8 waves x 2 ntiles -> Cout 256).
// Per group: coord table (144) -> bilinear A[16][576] bf16 in LDS ->
//            18 kslices x 2 WMMAs per wave.
// LDS A row stride 292 dwords (mult of 4; 292 mod 64 = 36 -> distinct banks).
// ======================================================================
#define S2_STRIDE_DW 292
__global__ __launch_bounds__(256) void deform_kernel(
    const float* __restrict__ x, const unsigned* __restrict__ pack2,
    const float* __restrict__ offs, float* __restrict__ out) {
    int bid = blockIdx.x;
    int wo0 = (bid & 7) * 16;
    int ho  = (bid >> 3) & 127;
    int n   = bid >> 10;
    int tid  = threadIdx.x;
    int lane = tid & 31;
    int wv   = tid >> 5;                   // 0..7

    __shared__ unsigned sA[16 * S2_STRIDE_DW];
    __shared__ float    sW[144 * 4];
    __shared__ int      sI[144 * 4];

    v8f acc0 = {}, acc1 = {};
    const float* offn = offs + (size_t)n * OCH * HOWO;

    for (int g = 0; g < GRP; ++g) {
        __syncthreads();
        // ---- coordinate / bilinear table: one entry per (pixel p, tap k) ----
        if (tid < 144) {
            int p = tid & 15, k = tid >> 4;
            int wo = wo0 + p;
            float dy = offn[(size_t)(g * 18 + k * 2 + 0) * HOWO + ho * WO + wo];
            float dx = offn[(size_t)(g * 18 + k * 2 + 1) * HOWO + ho * WO + wo];
            int ky = k / 3, kx = k - ky * 3;
            float y  = dy + (float)(ho + 2 * ky);
            float xx = dx + (float)(wo + 2 * kx);
            float y0f = floorf(y), x0f = floorf(xx);
            float ly = y - y0f, lx = xx - x0f;
            int y0 = (int)y0f, x0 = (int)x0f;
#pragma unroll
            for (int cnr = 0; cnr < 4; ++cnr) {
                int yi = y0 + (cnr >> 1);
                int xi = x0 + (cnr & 1);
                bool vld = (yi >= 0) & (yi < H_IN) & (xi >= 0) & (xi < W_IN);
                int yc = min(max(yi, 0), H_IN - 1);
                int xc = min(max(xi, 0), W_IN - 1);
                float wgt = ((cnr >> 1) ? ly : 1.0f - ly) * ((cnr & 1) ? lx : 1.0f - lx);
                sW[tid * 4 + cnr] = vld ? wgt : 0.0f;
                sI[tid * 4 + cnr] = yc * W_IN + xc;
            }
        }
        __syncthreads();
        // ---- bilinear gather fill of A: A[p][c*9+k] ----
        {
            const float* xg = x + (size_t)(n * CIN + g * CG) * PLANE;
            for (int e = tid; e < 16 * CG * KTAP; e += 256) {
                int p = e & 15;
                int c = (e >> 4) & 63;
                int k = e >> 10;
                int ent = (k * 16 + p) * 4;
                const float* xp = xg + (size_t)c * PLANE;
                float v = sW[ent + 0] * xp[sI[ent + 0]]
                        + sW[ent + 1] * xp[sI[ent + 1]]
                        + sW[ent + 2] * xp[sI[ent + 2]]
                        + sW[ent + 3] * xp[sI[ent + 3]];
                ((unsigned short*)sA)[p * (S2_STRIDE_DW * 2) + c * 9 + k] = f2bf(v);
            }
        }
        __syncthreads();
        // ---- GEMM: 18 kslices, 2 ntiles per wave ----
        {
            int row = lane & 15;
            int kbd = (lane >> 4) * 8;
            for (int ks = 0; ks < 18; ++ks) {
                const unsigned* ap = sA + row * S2_STRIDE_DW + ks * 16 + kbd;
                FragU fa; fa.lo = *(const uint4*)ap; fa.hi = *(const uint4*)(ap + 4);
                v16bf a = __builtin_bit_cast(v16bf, fa);
                const unsigned* bp0 = pack2 + (size_t)((g * 18 + ks) * 16 + wv * 2) * 256 + lane * 8;
                FragU f0; f0.lo = *(const uint4*)bp0;         f0.hi = *(const uint4*)(bp0 + 4);
                FragU f1; f1.lo = *(const uint4*)(bp0 + 256); f1.hi = *(const uint4*)(bp0 + 260);
                acc0 = __builtin_amdgcn_wmma_f32_16x16x32_bf16(
                    false, a, false, __builtin_bit_cast(v16bf, f0), (short)0, acc0, false, false);
                acc1 = __builtin_amdgcn_wmma_f32_16x16x32_bf16(
                    false, a, false, __builtin_bit_cast(v16bf, f1), (short)0, acc1, false, false);
            }
        }
    }

    // ---- store: lane's 8 accum VGPRs are contiguous in wo -> two float4 per acc ----
    int ncol  = lane & 15;
    int halfM = lane >> 4;
    V8FU u0; u0.v = acc0;
    V8FU u1; u1.v = acc1;
    size_t b0 = (((size_t)n * COUT + (wv * 2 + 0) * 16 + ncol) * HO + ho) * WO + wo0 + halfM * 8;
    size_t b1 = (((size_t)n * COUT + (wv * 2 + 1) * 16 + ncol) * HO + ho) * WO + wo0 + halfM * 8;
    *(float4*)(out + b0)     = u0.f4[0];
    *(float4*)(out + b0 + 4) = u0.f4[1];
    *(float4*)(out + b1)     = u1.f4[0];
    *(float4*)(out + b1 + 4) = u1.f4[1];
}

// ======================================================================
extern "C" void kernel_launch(void* const* d_in, const int* in_sizes, int n_in,
                              void* d_out, int out_size, void* d_ws, size_t ws_size,
                              hipStream_t stream) {
    const float* x    = (const float*)d_in[0];
    const float* offw = (const float*)d_in[1];
    const float* offb = (const float*)d_in[2];
    const float* dw   = (const float*)d_in[3];
    float* out = (float*)d_out;

    unsigned* wsu   = (unsigned*)d_ws;
    unsigned* pack1 = wsu + PACK1_DW;
    unsigned* pack2 = wsu + PACK2_DW;
    float*    offs  = (float*)(wsu + OFFS_DW);

    pack_offw_kernel<<<PACK1_SZ / 256, 256, 0, stream>>>(offw, pack1);
    pack_dw_kernel<<<PACK2_SZ / 256, 256, 0, stream>>>(dw, pack2);
    offset_conv_kernel<<<N_B * HO * (WO / 16), 160, 0, stream>>>(x, pack1, offb, offs);
    deform_kernel<<<N_B * HO * (WO / 16), 256, 0, stream>>>(x, pack2, offs, out);
}
